// MoE_78099685310874
// MI455X (gfx1250) — compile-verified, matrix operand order checked by reading, not run on hardware
//
#include <hip/hip_runtime.h>
#include <hip/hip_bf16.h>

// ---------------- Problem constants (match reference) ----------------
#define N_TOK   16384
#define D_MODEL 1024
#define D_INT   512
#define N_EXP   8

#define M_TILE  64      // tokens per block in the fused expert kernel
#define BLOCK   512     // 16 wave32s
#define WAVES   (BLOCK / 32)

// ---------------- Types for WMMA ----------------
typedef __attribute__((ext_vector_type(16))) __bf16 v16bf;
typedef __attribute__((ext_vector_type(8)))  float  v8f;

union Frag16 {          // one 16x32 bf16 A/B fragment = 32 bytes per lane
    uint4 u4[2];
    v16bf v;
};

#if __has_builtin(__builtin_amdgcn_global_load_async_to_lds_b128) && \
    __has_builtin(__builtin_amdgcn_s_wait_asynccnt)
#define HAVE_ASYNC_LDS 1
// exact operand type per hipcc diagnostic: pointer to 16B int vector
typedef int async_v4i __attribute__((vector_size(16)));
typedef __attribute__((address_space(1))) async_v4i* async_gptr;
typedef __attribute__((address_space(3))) async_v4i* async_lptr;
#else
#define HAVE_ASYNC_LDS 0
#endif

__device__ __forceinline__ unsigned short f2bf(float f) {
    unsigned int u = __float_as_uint(f);
    unsigned int r = (u + 0x7FFFu + ((u >> 16) & 1u)) >> 16;  // round-nearest-even
    return (unsigned short)r;
}

// fast silu: y * g * rcp(1 + exp(-g)); hardware v_rcp_f32 instead of IEEE divide
__device__ __forceinline__ float silu_gate(float y, float g) {
    return y * g * __builtin_amdgcn_rcpf(1.0f + __expf(-g));
}

// ---------------- f32 -> bf16 conversion (vectorized x4) ----------------
__global__ __launch_bounds__(256) void cvt_bf16_kernel(const float* __restrict__ in,
                                                       unsigned short* __restrict__ out,
                                                       int n4) {
    int i = blockIdx.x * 256 + threadIdx.x;
    if (i >= n4) return;
    float4 v = ((const float4*)in)[i];
    ushort4 o;
    o.x = f2bf(v.x); o.y = f2bf(v.y); o.z = f2bf(v.z); o.w = f2bf(v.w);
    ((ushort4*)out)[i] = o;
}

// ---------------- Gate: softmax -> top-2 -> renormalized coef ----------------
__global__ __launch_bounds__(256) void gate_kernel(const float* __restrict__ x,
                                                   const float* __restrict__ wg,
                                                   float* __restrict__ coef) {
    __shared__ float wgs[N_EXP * D_MODEL];   // 32 KB
    int tid = threadIdx.x;
    for (int i = tid; i < N_EXP * D_MODEL; i += 256) wgs[i] = wg[i];
    __syncthreads();

    int wave = tid >> 5, lane = tid & 31;
    int tok = blockIdx.x * 8 + wave;
    const float* xr = x + (size_t)tok * D_MODEL;

    float acc[N_EXP];
#pragma unroll
    for (int e = 0; e < N_EXP; ++e) acc[e] = 0.0f;

    for (int i = 0; i < D_MODEL / 32; ++i) {
        int idx = lane + 32 * i;
        float v = xr[idx];
#pragma unroll
        for (int e = 0; e < N_EXP; ++e) acc[e] += v * wgs[e * D_MODEL + idx];
    }
#pragma unroll
    for (int off = 16; off > 0; off >>= 1) {
#pragma unroll
        for (int e = 0; e < N_EXP; ++e) acc[e] += __shfl_xor(acc[e], off, 32);
    }
    float mx = acc[0]; int i1 = 0;
#pragma unroll
    for (int e = 1; e < N_EXP; ++e) if (acc[e] > mx) { mx = acc[e]; i1 = e; }
    float m2 = -3.0e38f; int i2 = 0;
#pragma unroll
    for (int e = 0; e < N_EXP; ++e)
        if (e != i1 && acc[e] > m2) { m2 = acc[e]; i2 = e; }
    float e2 = __expf(m2 - mx);
    float inv = __builtin_amdgcn_rcpf(1.0f + e2);
    if (lane < N_EXP) {
        float c = (lane == i1) ? inv : ((lane == i2) ? e2 * inv : 0.0f);
        coef[(size_t)tok * N_EXP + lane] = c;
    }
}

// ---------------- Fused expert MLP: (x @ fc1^T) -> silu-gate -> @ fc2^T ----------------
// Block = 64 tokens, 16 waves; x-tile staged in 128 KB LDS once, reused by all 8 experts.
// Per expert: stage A -> gated t (bf16) in 64 KB LDS; stage B -> coef-scaled global RMW.
__global__ __launch_bounds__(BLOCK) void moe_expert_kernel(
    const unsigned short* __restrict__ xbf,    // N_TOK x D_MODEL   (bf16)
    const unsigned short* __restrict__ fc1bf,  // N_EXP x 2*D_INT x D_MODEL
    const unsigned short* __restrict__ fc2bf,  // N_EXP x D_MODEL x D_INT
    const float* __restrict__ coef,            // N_TOK x N_EXP
    float* __restrict__ out)                   // N_TOK x D_MODEL
{
    __shared__ unsigned short xs[M_TILE * D_MODEL];   // 128 KB (16B-chunk XOR swizzle)
    __shared__ unsigned short ts[M_TILE * D_INT];     // 64 KB
    __shared__ float cs[M_TILE * N_EXP];              // 2 KB

    int tid  = threadIdx.x;
    int wave = tid >> 5, lane = tid & 31;
    int m0   = blockIdx.x * M_TILE;
    int n16  = lane & 15;          // column (B/D) or row (A) within 16
    int half = lane >> 4;          // K-half selector per ISA layout

    cs[tid] = coef[(size_t)m0 * N_EXP + tid];          // 512 = M_TILE*N_EXP

    // stage x tile: coalesced 16B per thread, chunk index XOR-swizzled by row
    {
        const uint4* xg = (const uint4*)(xbf + (size_t)m0 * D_MODEL);
        uint4* xs4 = (uint4*)xs;
        for (int i = tid; i < M_TILE * (D_MODEL / 8); i += BLOCK) {
            int row = i >> 7, ch = i & 127;            // 128 chunks/row
            int dst = row * 128 + (ch ^ (row & 7));
#if HAVE_ASYNC_LDS
            __builtin_amdgcn_global_load_async_to_lds_b128(
                (async_gptr)(xg + i), (async_lptr)(xs4 + dst), 0, 0);
#else
            xs4[dst] = xg[i];
#endif
        }
#if HAVE_ASYNC_LDS
        __builtin_amdgcn_s_wait_asynccnt(0);
#endif
    }
    __syncthreads();

    const uint4* xs4 = (const uint4*)xs;
    const uint4* ts4 = (const uint4*)ts;

    for (int e = 0; e < N_EXP; ++e) {
        // ---------------- Stage A: h = x @ fc1[e]^T, t = y * silu(g) ----------------
        const unsigned short* W1 = fc1bf + (size_t)e * (2 * D_INT) * D_MODEL;
        for (int j = 0; j < 2; ++j) {                  // 32 y-chunks over 16 waves
            int yc = wave * 2 + j;
            int ny = yc * 16 + n16;                    // fc1 row for y half
            int ng = D_INT + yc * 16 + n16;            // fc1 row for gate half
            const uint4* By0 = (const uint4*)(W1 + (size_t)ny * D_MODEL);
            const uint4* Bg0 = (const uint4*)(W1 + (size_t)ng * D_MODEL);

            v8f zero = {0.f,0.f,0.f,0.f,0.f,0.f,0.f,0.f};
            v8f accy[4], accg[4];
#pragma unroll
            for (int mt = 0; mt < 4; ++mt) { accy[mt] = zero; accg[mt] = zero; }

            for (int k0 = 0; k0 < D_MODEL; k0 += 32) {
                int cb = k0 >> 3;                      // 16B-chunk index of k0
                Frag16 by, bg;
                by.u4[0] = By0[cb + half];     by.u4[1] = By0[cb + 2 + half];
                bg.u4[0] = Bg0[cb + half];     bg.u4[1] = Bg0[cb + 2 + half];
                Frag16 a[4];
#pragma unroll
                for (int mt = 0; mt < 4; ++mt) {
                    int row = mt * 16 + n16;           // A-matrix row = lane&15
                    int base = row * 128;
                    a[mt].u4[0] = xs4[base + ((cb + half)     ^ (row & 7))];
                    a[mt].u4[1] = xs4[base + ((cb + 2 + half) ^ (row & 7))];
                }
#pragma unroll
                for (int mt = 0; mt < 4; ++mt) {
                    accy[mt] = __builtin_amdgcn_wmma_f32_16x16x32_bf16(
                        false, a[mt].v, false, by.v, (short)0, accy[mt], false, false);
                    accg[mt] = __builtin_amdgcn_wmma_f32_16x16x32_bf16(
                        false, a[mt].v, false, bg.v, (short)0, accg[mt], false, false);
                }
            }
            // silu-gate and park t in LDS (bf16, swizzled to match stage-B A reads)
#pragma unroll
            for (int mt = 0; mt < 4; ++mt) {
#pragma unroll
                for (int r = 0; r < 8; ++r) {
                    float t = silu_gate(accy[mt][r], accg[mt][r]);
                    int row = mt * 16 + r + 8 * half;  // D layout: M = r + 8*(lane>=16)
                    int col = yc * 16 + n16;
                    int ch = col >> 3, wi = col & 7;   // 64 chunks/row in ts
                    ts[row * D_INT + (((ch ^ (row & 7)) << 3) + wi)] = f2bf(t);
                }
            }
        }
        __syncthreads();

        // ---------------- Stage B: out += coef * (t @ fc2[e]^T) ----------------
        const unsigned short* W2 = fc2bf + (size_t)e * D_MODEL * D_INT;
        for (int j = 0; j < 2; ++j) {                  // 64 out-chunks over 16 waves
            int nc0 = wave * 4 + j * 2;                // two adjacent chunks per pass
            int nn0 = nc0 * 16 + n16;
            int nn1 = (nc0 + 1) * 16 + n16;
            const uint4* B0 = (const uint4*)(W2 + (size_t)nn0 * D_INT);
            const uint4* B1 = (const uint4*)(W2 + (size_t)nn1 * D_INT);

            v8f zero = {0.f,0.f,0.f,0.f,0.f,0.f,0.f,0.f};
            v8f acc0[4], acc1[4];
#pragma unroll
            for (int mt = 0; mt < 4; ++mt) { acc0[mt] = zero; acc1[mt] = zero; }

            for (int k0 = 0; k0 < D_INT; k0 += 32) {
                int cb = k0 >> 3;
                Frag16 b0, b1;
                b0.u4[0] = B0[cb + half]; b0.u4[1] = B0[cb + 2 + half];
                b1.u4[0] = B1[cb + half]; b1.u4[1] = B1[cb + 2 + half];
                Frag16 a[4];
#pragma unroll
                for (int mt = 0; mt < 4; ++mt) {
                    int row = mt * 16 + n16;
                    int base = row * 64;               // 64 chunks/row in ts
                    a[mt].u4[0] = ts4[base + ((cb + half)     ^ (row & 7))];
                    a[mt].u4[1] = ts4[base + ((cb + 2 + half) ^ (row & 7))];
                }
#pragma unroll
                for (int mt = 0; mt < 4; ++mt) {
                    acc0[mt] = __builtin_amdgcn_wmma_f32_16x16x32_bf16(
                        false, a[mt].v, false, b0.v, (short)0, acc0[mt], false, false);
                    acc1[mt] = __builtin_amdgcn_wmma_f32_16x16x32_bf16(
                        false, a[mt].v, false, b1.v, (short)0, acc1[mt], false, false);
                }
            }
#pragma unroll
            for (int mt = 0; mt < 4; ++mt) {
#pragma unroll
                for (int r = 0; r < 8; ++r) {
                    int row = mt * 16 + r + 8 * half;
                    float c = cs[row * N_EXP + e];
                    float v0 = c * acc0[mt][r];
                    float v1 = c * acc1[mt][r];
                    float* op0 = out + (size_t)(m0 + row) * D_MODEL + nn0;
                    float* op1 = out + (size_t)(m0 + row) * D_MODEL + nn1;
                    if (e == 0) { *op0 = v0;  *op1 = v1;  }   // first expert initializes
                    else        { *op0 += v0; *op1 += v1; }   // block owns rows: race-free
                }
            }
        }
        __syncthreads();   // protect ts before next expert's stage A
    }
}

// ---------------- Host launcher ----------------
extern "C" void kernel_launch(void* const* d_in, const int* in_sizes, int n_in,
                              void* d_out, int out_size, void* d_ws, size_t ws_size,
                              hipStream_t stream) {
    const float* x   = (const float*)d_in[0];   // (N_TOK, D_MODEL)
    const float* wg  = (const float*)d_in[1];   // (N_EXP, D_MODEL)
    const float* fc1 = (const float*)d_in[2];   // (N_EXP, 2*D_INT, D_MODEL)
    const float* fc2 = (const float*)d_in[3];   // (N_EXP, D_MODEL, D_INT)
    // d_in[4] = top_k (fixed at 2 in this kernel)
    float* out = (float*)d_out;

    // workspace layout (bf16 copies + dense coef): ~56.5 MB total
    size_t n_x   = (size_t)N_TOK * D_MODEL;
    size_t n_fc1 = (size_t)N_EXP * 2 * D_INT * D_MODEL;
    size_t n_fc2 = (size_t)N_EXP * D_MODEL * D_INT;
    unsigned short* xbf   = (unsigned short*)d_ws;
    unsigned short* fc1bf = xbf + n_x;
    unsigned short* fc2bf = fc1bf + n_fc1;
    float*          coef  = (float*)(fc2bf + n_fc2);   // byte offset is 4B-aligned

    int n4;
    n4 = (int)(n_x / 4);
    cvt_bf16_kernel<<<(n4 + 255) / 256, 256, 0, stream>>>(x, xbf, n4);
    n4 = (int)(n_fc1 / 4);
    cvt_bf16_kernel<<<(n4 + 255) / 256, 256, 0, stream>>>(fc1, fc1bf, n4);
    n4 = (int)(n_fc2 / 4);
    cvt_bf16_kernel<<<(n4 + 255) / 256, 256, 0, stream>>>(fc2, fc2bf, n4);

    gate_kernel<<<N_TOK / 8, 256, 0, stream>>>(x, wg, coef);

    moe_expert_kernel<<<N_TOK / M_TILE, BLOCK, 0, stream>>>(xbf, fc1bf, fc2bf, coef, out);
}
